// AttnBlock_69295002353733
// MI455X (gfx1250) — matrix operand order repaired and checked by reading, not compiled, not run on hardware
//
#include <hip/hip_runtime.h>

typedef _Float16 half_t;
typedef __attribute__((ext_vector_type(16))) _Float16 v16h;
typedef __attribute__((ext_vector_type(8)))  _Float16 v8h;
typedef __attribute__((ext_vector_type(8)))  float    v8f;
typedef __attribute__((ext_vector_type(4)))  unsigned int u32x4;
typedef __attribute__((ext_vector_type(8)))  int i32x8;
typedef __attribute__((ext_vector_type(4)))  int i32x4;

union V16 { v16h v; v8h h[2]; };

// A operand (16x32 f16): rowm points to row M, contiguous over K.
// Lanes 0-15 take K chunks {0..7,16..23}, lanes 16-31 take {8..15,24..31} (ISA 7.12.2).
__device__ __forceinline__ v16h load_a(const half_t* rowm, int lane) {
  int kb = (lane >= 16) ? 8 : 0;
  V16 r;
  r.h[0] = *(const v8h*)(rowm + kb);
  r.h[1] = *(const v8h*)(rowm + kb + 16);
  return r.v;
}
// B operand (32x16 f16): coln points to column N storage, contiguous over K.
// Lanes 0-15 take K chunks {0..7,16..23}, lanes 16-31 take {8..15,24..31} (same as A).
__device__ __forceinline__ v16h load_b(const half_t* coln, int lane) {
  int kb = (lane >= 16) ? 8 : 0;
  V16 r;
  r.h[0] = *(const v8h*)(coln + kb);
  r.h[1] = *(const v8h*)(coln + kb + 16);
  return r.v;
}

__device__ __forceinline__ v8f wmma16(v16h a, v16h b, v8f c) {
  return __builtin_amdgcn_wmma_f32_16x16x32_f16(false, a, false, b, (short)0, c, false, false);
}

__device__ __forceinline__ v8f vzero8() {
  v8f z = {0.f, 0.f, 0.f, 0.f, 0.f, 0.f, 0.f, 0.f};
  return z;
}

#define B_ 8
#define C_ 512
#define L_ 2048
#define GC_ 64   // channels per group

// ---- TDM: 2D tile load (tile_rows x 512 halves, row-major, stride 512) -> LDS ----
// D# per ISA ch.8: group0 = {flags, lds_addr, gaddr lo, gaddr hi | type=2}
// group1 = {data_size=2B, tensor_dim0=512, tensor_dim1=2048, tile_dim0=512,
//           tile_dim1=tile_rows, tensor_dim0_stride=512}
__device__ __forceinline__ void tdm_load_tile(unsigned lds_off, const half_t* gptr,
                                              unsigned tile_rows) {
  unsigned long long ga = (unsigned long long)(uintptr_t)gptr;
  u32x4 g0;
  g0[0] = 1u;                                                  // count=1, user descriptor
  g0[1] = lds_off;                                             // lds_addr (bytes)
  g0[2] = (unsigned)(ga & 0xFFFFFFFFull);                      // global_addr[31:0]
  g0[3] = (unsigned)((ga >> 32) & 0x1FFFFFFull) | (2u << 30);  // addr[56:32] | type=2
  i32x8 g1;
  g1[0] = (int)(1u << 16);        // workgroup_mask=0 | data_size=1 (2 bytes)
  g1[1] = (int)(512u << 16);      // tensor_dim0[15:0]=512 in [31:16]
  g1[2] = (int)(2048u << 16);     // tensor_dim0[31:16]=0 | tensor_dim1[15:0]=2048
  g1[3] = (int)(512u << 16);      // tensor_dim1[31:16]=0 | tile_dim0=512
  g1[4] = (int)tile_rows;         // tile_dim1 | tile_dim2=0
  g1[5] = 512;                    // tensor_dim0_stride[31:0]=512
  g1[6] = 0;                      // dim0_stride hi | dim1_stride lo
  g1[7] = 0;
  i32x4 z4 = {0, 0, 0, 0};
  i32x8 z8 = {0, 0, 0, 0, 0, 0, 0, 0};
  // amdgpu-toolchain (clang-23) 6-arg form: (g0, g1, g2, g3, g4, cpol)
  __builtin_amdgcn_tensor_load_to_lds(g0, g1, z4, z4, z8, 0);
}

// ---------------- Kernel 1: GroupNorm -> hT[b][i][c] (f16, transposed) ----------------
__global__ void gnorm_kernel(const float* __restrict__ x, const float* __restrict__ gamma,
                             const float* __restrict__ beta, half_t* __restrict__ hT) {
  int bg = blockIdx.x;
  int b = bg >> 3, g = bg & 7;
  const float* xg = x + ((size_t)b * C_ + (size_t)g * GC_) * L_;  // contiguous 64*2048 region
  int tid = threadIdx.x;
  __shared__ float red[256], red2[256];
  float s = 0.f, s2 = 0.f;
  for (int idx = tid; idx < GC_ * L_; idx += 256) {
    float v = xg[idx];
    s += v; s2 += v * v;
  }
  red[tid] = s; red2[tid] = s2;
  __syncthreads();
  for (int off = 128; off > 0; off >>= 1) {
    if (tid < off) { red[tid] += red[tid + off]; red2[tid] += red2[tid + off]; }
    __syncthreads();
  }
  const float inv_n = 1.0f / (float)(GC_ * L_);
  float mean = red[0] * inv_n;
  float var  = red2[0] * inv_n - mean * mean;
  float rstd = rsqrtf(var + 1e-6f);
  half_t* hb = hT + (size_t)b * L_ * C_;
  for (int idx = tid; idx < GC_ * L_; idx += 256) {
    int c = idx >> 11;            // / L_
    int i = idx & (L_ - 1);
    int ch = g * GC_ + c;
    float v = (xg[idx] - mean) * rstd * gamma[ch] + beta[ch];
    hb[(size_t)i * C_ + ch] = (half_t)v;
  }
}

// ---------------- Kernel 2: convert 4 weight matrices f32 -> f16 (row-major) ----------------
__global__ void wcvt_kernel(const float* __restrict__ wq, const float* __restrict__ wk,
                            const float* __restrict__ wv, const float* __restrict__ wo,
                            half_t* __restrict__ dst) {
  int idx = blockIdx.x * 256 + threadIdx.x;    // 4 * 262144 total
  int m = idx >> 18;
  int off = idx & 262143;
  const float* s = (m == 0) ? wq : (m == 1) ? wk : (m == 2) ? wv : wo;
  dst[idx] = (half_t)s[off];
}

// ---------------- Kernel 3: projection GEMM: outT[b][i][n] = sum_k hT[b][i][k]*W[n][k] + bias[n]
__global__ void proj_kernel(const half_t* __restrict__ hT, const half_t* __restrict__ W,
                            const float* __restrict__ bias, half_t* __restrict__ outT) {
  int b = blockIdx.z;
  int i0 = blockIdx.x * 128;
  int n0 = blockIdx.y * 64;
  int tid = threadIdx.x;
  int wv_ = tid >> 5, lane = tid & 31;
  int m_row = i0 + wv_ * 16 + (lane & 15);
  const half_t* arow = hT + ((size_t)b * L_ + m_row) * C_;
  v8f acc[4];
  int ncol[4];
#pragma unroll
  for (int t = 0; t < 4; ++t) { acc[t] = vzero8(); ncol[t] = n0 + t * 16 + (lane & 15); }
  for (int k0 = 0; k0 < C_; k0 += 32) {
    v16h a = load_a(arow + k0, lane);
#pragma unroll
    for (int t = 0; t < 4; ++t) {
      v16h bb = load_b(W + (size_t)ncol[t] * C_ + k0, lane);
      acc[t] = wmma16(a, bb, acc[t]);
    }
  }
  half_t* ob = outT + (size_t)b * L_ * C_;
  int hi = (lane >= 16) ? 8 : 0;
#pragma unroll
  for (int t = 0; t < 4; ++t) {
    float bs = bias[ncol[t]];
#pragma unroll
    for (int k = 0; k < 8; ++k) {
      int i = i0 + wv_ * 16 + hi + k;
      ob[(size_t)i * C_ + ncol[t]] = (half_t)(acc[t][k] + bs);
    }
  }
}

// ---------------- Kernel 4: flash attention over 64-query tiles ----------------
// LDS map (bytes):
//   Sbuf  f32[64*64]   @ 0       (16384)
//   Pbuf  f16[64*64]   @ 16384   ( 8192)
//   Vlds  f16[512*64]  @ 24576   (65536)  transposed V tile (staged by VALU)
//   Klds  f16[64*512]  @ 90112   (65536)  row-major K tile  (staged by TDM)
//   Qlds  f16[64*512]  @ 155648  (65536)  row-major Q tile  (staged by TDM, once)
//   stats              @ 221184  ( 1024)
#define SOFF_P 16384
#define SOFF_V 24576
#define SOFF_K 90112
#define SOFF_Q 155648
#define SOFF_ST 221184
#define ATTN_LDS_BYTES (SOFF_ST + 1024)
__global__ void attn_kernel(const half_t* __restrict__ qT, const half_t* __restrict__ kT,
                            const half_t* __restrict__ vT, half_t* __restrict__ aT) {
  extern __shared__ char smem[];
  float*  Sbuf = (float*)smem;
  half_t* Pbuf = (half_t*)(smem + SOFF_P);
  half_t* Vlds = (half_t*)(smem + SOFF_V);
  half_t* Klds = (half_t*)(smem + SOFF_K);
  half_t* Qlds = (half_t*)(smem + SOFF_Q);
  float*  mrow = (float*)(smem + SOFF_ST);
  float*  lrow = mrow + 64;
  float*  srow = lrow + 64;
  int b = blockIdx.y;
  int i0 = blockIdx.x * 64;
  int tid = threadIdx.x;
  int wv_ = tid >> 5, lane = tid & 31;
  const half_t* qb = qT + (size_t)b * L_ * C_;
  const half_t* kb = kT + (size_t)b * L_ * C_;
  const half_t* vb = vT + (size_t)b * L_ * C_;
  // Q tile for this block: one TDM copy, reused across all 32 j-steps.
  if (wv_ == 0) tdm_load_tile(SOFF_Q, qb + (size_t)i0 * C_, 64);
  if (tid < 64) { mrow[tid] = -1e30f; lrow[tid] = 0.f; }
  // each wave owns output c-slice [wv_*64, +64), all 4 i-subtiles: 16 accumulators
  v8f oacc[4][4];
#pragma unroll
  for (int mi = 0; mi < 4; ++mi)
#pragma unroll
    for (int ct = 0; ct < 4; ++ct) oacc[mi][ct] = vzero8();
  const float rsC = 0.044194173824159216f;  // 512^-0.5
  if (wv_ == 0) __builtin_amdgcn_s_wait_tensorcnt(0);
  __syncthreads();

  for (int j0 = 0; j0 < L_; j0 += 64) {
    // async: TDM pulls K tile row-major into Klds while all threads transpose-stage V
    if (wv_ == 0) tdm_load_tile(SOFF_K, kb + (size_t)j0 * C_, 64);
    // stage V tile transposed: Vlds[c*64 + jl] = vT[j0+jl][c]
#pragma unroll 4
    for (int u = 0; u < 128; ++u) {
      int flat = tid + u * 256;
      int c = flat & 511, jl = flat >> 9;
      Vlds[c * 64 + jl] = vb[(size_t)(j0 + jl) * C_ + c];
    }
    if (wv_ == 0) __builtin_amdgcn_s_wait_tensorcnt(0);
    __syncthreads();
    // ---- S = (Q^T K) * rsC : 16 16x16 tiles, 2 per wave (same mi -> shared A), LDS-fed ----
    {
      int t0 = wv_ * 2;
      int mi = t0 >> 2, nj = t0 & 3;
      const half_t* arow = Qlds + (size_t)(mi * 16 + (lane & 15)) * C_;
      const half_t* bc0 = Klds + (size_t)(nj * 16 + (lane & 15)) * C_;
      const half_t* bc1 = bc0 + (size_t)16 * C_;
      v8f s0 = vzero8(), s1 = vzero8();
      for (int k0 = 0; k0 < C_; k0 += 32) {
        v16h a = load_a(arow + k0, lane);
        s0 = wmma16(a, load_b(bc0 + k0, lane), s0);
        s1 = wmma16(a, load_b(bc1 + k0, lane), s1);
      }
      int hi = (lane >= 16) ? 8 : 0;
#pragma unroll
      for (int k = 0; k < 8; ++k) {
        int r = mi * 16 + hi + k;
        Sbuf[r * 64 + nj * 16 + (lane & 15)]        = s0[k] * rsC;
        Sbuf[r * 64 + (nj + 1) * 16 + (lane & 15)]  = s1[k] * rsC;
      }
    }
    __syncthreads();
    // ---- online softmax: 4 threads per row ----
    {
      int r = tid >> 2, q = tid & 3;
      float* srp = Sbuf + r * 64 + q * 16;
      float mx = -1e30f;
#pragma unroll
      for (int e = 0; e < 16; ++e) mx = fmaxf(mx, srp[e]);
      mx = fmaxf(mx, __shfl_xor(mx, 1, 4));
      mx = fmaxf(mx, __shfl_xor(mx, 2, 4));
      float mold = mrow[r];
      float mnew = fmaxf(mold, mx);
      float sum = 0.f;
      half_t* prp = Pbuf + r * 64 + q * 16;
#pragma unroll
      for (int e = 0; e < 16; ++e) {
        float p = __expf(srp[e] - mnew);
        sum += p;
        prp[e] = (half_t)p;
      }
      sum += __shfl_xor(sum, 1, 4);
      sum += __shfl_xor(sum, 2, 4);
      if (q == 0) {
        float sc = __expf(mold - mnew);
        srow[r] = sc;
        lrow[r] = lrow[r] * sc + sum;
        mrow[r] = mnew;
      }
    }
    __syncthreads();
    // ---- rescale O accumulators, then O^T += P * V^T ----
    {
      int hi = (lane >= 16) ? 8 : 0;
#pragma unroll
      for (int mi = 0; mi < 4; ++mi) {
        float sc[8];
#pragma unroll
        for (int k = 0; k < 8; ++k) sc[k] = srow[mi * 16 + hi + k];
#pragma unroll
        for (int ct = 0; ct < 4; ++ct)
#pragma unroll
          for (int k = 0; k < 8; ++k) oacc[mi][ct][k] *= sc[k];
      }
#pragma unroll
      for (int ks = 0; ks < 64; ks += 32) {
        v16h av[4];
#pragma unroll
        for (int mi = 0; mi < 4; ++mi)
          av[mi] = load_a(Pbuf + (mi * 16 + (lane & 15)) * 64 + ks, lane);
#pragma unroll
        for (int ct = 0; ct < 4; ++ct) {
          int c = wv_ * 64 + ct * 16 + (lane & 15);
          v16h bb = load_b(Vlds + c * 64 + ks, lane);
#pragma unroll
          for (int mi = 0; mi < 4; ++mi)
            oacc[mi][ct] = wmma16(av[mi], bb, oacc[mi][ct]);
        }
      }
    }
    __syncthreads();
  }
  // ---- normalize by 1/l and write aT[b][i][c] f16 ----
  half_t* ab = aT + (size_t)b * L_ * C_;
  int hi = (lane >= 16) ? 8 : 0;
#pragma unroll
  for (int mi = 0; mi < 4; ++mi) {
    float rl[8];
#pragma unroll
    for (int k = 0; k < 8; ++k) rl[k] = 1.0f / lrow[mi * 16 + hi + k];
#pragma unroll
    for (int ct = 0; ct < 4; ++ct) {
      int c = wv_ * 64 + ct * 16 + (lane & 15);
#pragma unroll
      for (int k = 0; k < 8; ++k) {
        int i = i0 + mi * 16 + hi + k;
        ab[(size_t)i * C_ + c] = (half_t)(oacc[mi][ct][k] * rl[k]);
      }
    }
  }
}

// ---------------- Kernel 5: output projection + residual (f32 out) ----------------
__global__ void oproj_kernel(const float* __restrict__ x, const half_t* __restrict__ aT,
                             const half_t* __restrict__ Wo, const float* __restrict__ bo,
                             float* __restrict__ out) {
  int b = blockIdx.z;
  int i0 = blockIdx.x * 128;
  int n0 = blockIdx.y * 64;
  int tid = threadIdx.x;
  int wv_ = tid >> 5, lane = tid & 31;
  int m_row = i0 + wv_ * 16 + (lane & 15);
  const half_t* arow = aT + ((size_t)b * L_ + m_row) * C_;
  v8f acc[4];
#pragma unroll
  for (int t = 0; t < 4; ++t) acc[t] = vzero8();
  for (int k0 = 0; k0 < C_; k0 += 32) {
    v16h a = load_a(arow + k0, lane);
#pragma unroll
    for (int t = 0; t < 4; ++t) {
      int n = n0 + t * 16 + (lane & 15);
      v16h bb = load_b(Wo + (size_t)n * C_ + k0, lane);
      acc[t] = wmma16(a, bb, acc[t]);
    }
  }
  int hi = (lane >= 16) ? 8 : 0;
  const float* xb = x + (size_t)b * C_ * L_;
  float* ob = out + (size_t)b * C_ * L_;
#pragma unroll
  for (int t = 0; t < 4; ++t) {
    int cch = n0 + t * 16 + (lane & 15);
    float bs = bo[cch];
#pragma unroll
    for (int k = 0; k < 8; ++k) {
      int i = i0 + wv_ * 16 + hi + k;           // consecutive i per k -> contiguous stores
      size_t idx = (size_t)cch * L_ + i;
      ob[idx] = xb[idx] + acc[t][k] + bs;
    }
  }
}

extern "C" void kernel_launch(void* const* d_in, const int* in_sizes, int n_in,
                              void* d_out, int out_size, void* d_ws, size_t ws_size,
                              hipStream_t stream) {
  (void)in_sizes; (void)n_in; (void)out_size; (void)ws_size;
  const float* x     = (const float*)d_in[0];
  const float* gamma = (const float*)d_in[1];
  const float* beta  = (const float*)d_in[2];
  const float* wq    = (const float*)d_in[3];
  const float* bq    = (const float*)d_in[4];
  const float* wk    = (const float*)d_in[5];
  const float* bk    = (const float*)d_in[6];
  const float* wv    = (const float*)d_in[7];
  const float* bv    = (const float*)d_in[8];
  const float* wo    = (const float*)d_in[9];
  const float* bo    = (const float*)d_in[10];
  float* out = (float*)d_out;
  char* ws = (char*)d_ws;

  const size_t SZ = (size_t)B_ * L_ * C_ * sizeof(half_t);  // 16 MB per f16 activation buffer
  half_t* hT  = (half_t*)(ws);
  half_t* qT  = (half_t*)(ws + 1 * SZ);
  half_t* kTb = (half_t*)(ws + 2 * SZ);
  half_t* vTb = (half_t*)(ws + 3 * SZ);
  half_t* aT  = (half_t*)(ws + 4 * SZ);
  half_t* wf  = (half_t*)(ws + 5 * SZ);                     // 4 x 512x512 f16 = 2 MB

  wcvt_kernel <<<4096, 256, 0, stream>>>(wq, wk, wv, wo, wf);
  gnorm_kernel<<<B_ * 8, 256, 0, stream>>>(x, gamma, beta, hT);
  proj_kernel <<<dim3(L_ / 128, C_ / 64, B_), 256, 0, stream>>>(hT, wf + 0 * 262144, bq, qT);
  proj_kernel <<<dim3(L_ / 128, C_ / 64, B_), 256, 0, stream>>>(hT, wf + 1 * 262144, bk, kTb);
  proj_kernel <<<dim3(L_ / 128, C_ / 64, B_), 256, 0, stream>>>(hT, wf + 2 * 262144, bv, vTb);
  attn_kernel <<<dim3(L_ / 64, B_), 256, ATTN_LDS_BYTES, stream>>>(qT, kTb, vTb, aT);
  oproj_kernel<<<dim3(L_ / 128, C_ / 64, B_), 256, 0, stream>>>(x, aT, wf + 3 * 262144, bo, out);
}
